// MultiHeadSelfAttention_40922448396700
// MI455X (gfx1250) — matrix operand order repaired and checked by reading, not compiled, not run on hardware
//
#include <hip/hip_runtime.h>
#include <hip/hip_bf16.h>

typedef __attribute__((ext_vector_type(16))) _Float16 v16h;
typedef __attribute__((ext_vector_type(8)))  _Float16 v8h;
typedef __attribute__((ext_vector_type(4)))  _Float16 v4h;
typedef __attribute__((ext_vector_type(8)))  float    v8f;

#define Bb 4
#define Ss 2048
#define Dd 1024
#define Hh 64
#define NROWS (Bb*Ss)   /* 8192 */
#define LDK 72          /* padded LDS stride in halves (multiple of 8 -> 16B aligned rows) */

__device__ __forceinline__ v8f wmma16(v16h a, v16h b, v8f c){
  // D = A(16x32 f16) x B(32x16 f16) + C(16x16 f32)
  return __builtin_amdgcn_wmma_f32_16x16x32_f16(false, a, false, b, (short)0, c, false, false);
}

// A-fragment (16x32, MxK). ISA layout: lanes 0-15 -> M, halves j: k = j%8 + 16*(j>=8) + 8*(lane>=16)
__device__ __forceinline__ v16h load_afrag(const _Float16* p, int ld, int row, int k0, int lane){
  const int m = lane & 15, g = lane >> 4;
  const _Float16* q = p + (size_t)(row + m) * ld + k0 + g * 8;
  union { v16h v; v8h h[2]; } u;
  u.h[0] = *(const v8h*)(q);
  u.h[1] = *(const v8h*)(q + 16);
  return u.v;
}

// B-fragment (32x16, KxN) read from a K-contiguous (transposed) matrix: row index = n, halves = k.
// ISA layout: lane n = lane%16, k = j%8 + 16*(j>=8) + 8*(lane>=16)
__device__ __forceinline__ v16h load_bfrag(const _Float16* p, int ld, int col, int k0, int lane){
  const int n = lane & 15, g = lane >> 4;
  const _Float16* q = p + (size_t)(col + n) * ld + k0 + g * 8;
  union { v16h v; v8h h[2]; } u;
  u.h[0] = *(const v8h*)(q);
  u.h[1] = *(const v8h*)(q + 16);
  return u.v;
}

// ---------------- prep kernels ----------------

__global__ void cvt_x_kernel(const float* __restrict__ x, _Float16* __restrict__ xh){
  const size_t stride = (size_t)gridDim.x * blockDim.x;
  const size_t n4 = (size_t)NROWS * Dd / 4;
  for (size_t i = (size_t)blockIdx.x * blockDim.x + threadIdx.x; i < n4; i += stride){
    const float4 f = ((const float4*)x)[i];
    v4h o; o[0] = (_Float16)f.x; o[1] = (_Float16)f.y; o[2] = (_Float16)f.z; o[3] = (_Float16)f.w;
    ((v4h*)xh)[i] = o;
  }
}

// wt3: [3][64][1024] f16 = transposed Wq/Wk/Wv.  wsum_t: [1024][64] f16, wsum_t[e][j] = sum_h Wproj[h*64+j, e]
__global__ void prep_w_kernel(const float* __restrict__ Wq, const float* __restrict__ Wk,
                              const float* __restrict__ Wv, const float* __restrict__ Wp,
                              _Float16* __restrict__ wt3, _Float16* __restrict__ wsum_t){
  const int i = blockIdx.x * blockDim.x + threadIdx.x;   // 65536 threads
  if (i < Hh * Dd){
    const int nn = i >> 10, k = i & 1023;                // wt3[n][k] = W[k][n]
    wt3[i]               = (_Float16)Wq[(size_t)k * Hh + nn];
    wt3[Hh*Dd + i]       = (_Float16)Wk[(size_t)k * Hh + nn];
    wt3[2*Hh*Dd + i]     = (_Float16)Wv[(size_t)k * Hh + nn];
    const int e = i >> 6, j = i & 63;
    float s = 0.f;
#pragma unroll
    for (int h = 0; h < 16; ++h) s += Wp[(size_t)(h * 64 + j) * Dd + e];
    wsum_t[i] = (_Float16)s;
  }
}

// ---------------- QKV projection: [8192,1024] x [1024,64] x3 ----------------

__global__ __launch_bounds__(256) void qkv_kernel(const _Float16* __restrict__ xh,
                                                  const _Float16* __restrict__ wt3,
                                                  _Float16* __restrict__ qkv){
  const int lane = threadIdx.x & 31;
  const int wave = threadIdx.x >> 5;
  const int gw = blockIdx.x * 8 + wave;       // 512 m-tiles * 12 (3 mats * 4 n-tiles) = 6144 waves
  const int mt = gw / 12;
  const int rem = gw % 12;
  const int which = rem >> 2;
  const int nt = rem & 3;
  const int row = mt * 16;
  const _Float16* wtp = wt3 + (size_t)which * Hh * Dd;
  _Float16* outp = qkv + (size_t)which * NROWS * Hh;

  v8f c = {};
#pragma unroll 4
  for (int ks = 0; ks < Dd / 32; ++ks){
    v16h a  = load_afrag(xh,  Dd, row,      ks * 32, lane);
    v16h bf = load_bfrag(wtp, Dd, nt * 16,  ks * 32, lane);
    c = wmma16(a, bf, c);
  }
  const int n = lane & 15, g = lane >> 4;
#pragma unroll
  for (int r = 0; r < 8; ++r)
    outp[(size_t)(row + r + g * 8) * Hh + nt * 16 + n] = (_Float16)c[r];
}

// ---------------- flash attention, causal, one 64-dim head ----------------

__global__ __launch_bounds__(128) void attn_kernel(const _Float16* __restrict__ qh,
                                                   const _Float16* __restrict__ kh,
                                                   const _Float16* __restrict__ vh,
                                                   _Float16* __restrict__ ctxh){
  __shared__ __align__(16) _Float16 Kt[64 * LDK];       // K tile, row-major [key][dim]
  __shared__ __align__(16) _Float16 Vt[64 * LDK];       // V tile transposed [dim][key]
  __shared__ __align__(16) _Float16 Pb[4][16 * LDK];    // per-wave P staging (C-layout -> A-frag)

  const int tid  = threadIdx.x;
  const int lane = tid & 31;
  const int wave = tid >> 5;              // 4 waves, 16 q-rows each
  const int n    = lane & 15;
  const int g    = lane >> 4;

  const int qb   = blockIdx.x & 31;       // 32 q-blocks of 64 rows per batch
  const int b    = blockIdx.x >> 5;
  const int rowQ = b * Ss + qb * 64 + wave * 16;
  const int qpos = qb * 64 + wave * 16;   // sequence position of first q row
  const int nkt  = qb + 1;                // causal: k-tiles 0..qb

  const v16h qf0 = load_afrag(qh, Hh, rowQ, 0,  lane);
  const v16h qf1 = load_afrag(qh, Hh, rowQ, 32, lane);

  const unsigned long long kbase = (unsigned long long)(uintptr_t)kh;

  float mrow[8], lrow[8];
  v8f acc[4];
#pragma unroll
  for (int r = 0; r < 8; ++r){ mrow[r] = -1e30f; lrow[r] = 0.f; }
#pragma unroll
  for (int nc = 0; nc < 4; ++nc) acc[nc] = {};

  for (int kt = 0; kt < nkt; ++kt){
    const int rowK = b * Ss + kt * 64;
    __syncthreads();   // previous tile fully consumed before async writes land in Kt

    // stage K tile: CDNA5 async memory->LDS copy (no VGPR round-trip), tracked by ASYNCcnt.
    // GVS mode: addr = SADDR(base) + VADDR(byte offset); vdst VGPR = LDS byte offset.
    for (int idx = tid; idx < 64 * 8; idx += 128){
      const int row = idx >> 3, seg = idx & 7;                  // 8 halves (16B) per transfer
      const unsigned int voff = (unsigned int)((((unsigned)(rowK + row)) * Hh + seg * 8) * 2);
      const unsigned int ldst = (unsigned int)(uintptr_t)&Kt[row * LDK + seg * 8];
      asm volatile("global_load_async_to_lds_b128 %0, %1, %2"
                   :: "v"(ldst), "v"(voff), "s"(kbase) : "memory");
    }
    // stage V transposed (manual: transpose cannot use the straight async copy)
    for (int idx = tid; idx < 64 * 8; idx += 128){
      const int key = idx >> 3, seg = idx & 7;
      v8h d = *(const v8h*)(vh + (size_t)(rowK + key) * Hh + seg * 8);
#pragma unroll
      for (int e = 0; e < 8; ++e) Vt[(seg * 8 + e) * LDK + key] = d[e];
    }
    if (kt + 1 < nkt){  // prefetch next tile into L2 (global_prefetch_b8)
      __builtin_prefetch(kh + (size_t)(rowK + 64) * Hh + tid * 4, 0, 0);
      __builtin_prefetch(vh + (size_t)(rowK + 64) * Hh + tid * 4, 0, 0);
    }
    asm volatile("s_wait_asynccnt 0" ::: "memory");   // async K-tile writes visible in LDS
    __syncthreads();

    if (kt * 64 <= qpos + 15){            // wave-uniform: EXEC stays all-ones for WMMA
      // S = Q K^T (4 sub-tiles of 16 keys), scale + causal mask
      v8f sc[4];
#pragma unroll
      for (int sub = 0; sub < 4; ++sub){
        v8f s = {};
        s = wmma16(qf0, load_bfrag(Kt, LDK, sub * 16, 0,  lane), s);
        s = wmma16(qf1, load_bfrag(Kt, LDK, sub * 16, 32, lane), s);
        const int key = kt * 64 + sub * 16 + n;
#pragma unroll
        for (int r = 0; r < 8; ++r){
          float v = s[r] * 0.125f;                 // 1/sqrt(64)
          if (key > qpos + r + g * 8) v = -1e30f;  // causal
          s[r] = v;
        }
        sc[sub] = s;
      }
      // online softmax: row stats live across 16 lanes of each half-wave
      float mnew[8], corr[8];
#pragma unroll
      for (int r = 0; r < 8; ++r){
        float m4 = fmaxf(fmaxf(sc[0][r], sc[1][r]), fmaxf(sc[2][r], sc[3][r]));
#pragma unroll
        for (int off = 1; off < 16; off <<= 1)
          m4 = fmaxf(m4, __shfl_xor(m4, off, 32));
        mnew[r] = fmaxf(mrow[r], m4);
        corr[r] = __expf(mrow[r] - mnew[r]);
        mrow[r] = mnew[r];
      }
      _Float16* Pw = Pb[wave];
#pragma unroll
      for (int sub = 0; sub < 4; ++sub){
#pragma unroll
        for (int r = 0; r < 8; ++r){
          const float p = __expf(sc[sub][r] - mnew[r]);
          sc[sub][r] = p;
          Pw[(r + g * 8) * LDK + sub * 16 + n] = (_Float16)p;
        }
      }
#pragma unroll
      for (int r = 0; r < 8; ++r){
        float s4 = sc[0][r] + sc[1][r] + sc[2][r] + sc[3][r];
#pragma unroll
        for (int off = 1; off < 16; off <<= 1)
          s4 += __shfl_xor(s4, off, 32);
        lrow[r] = lrow[r] * corr[r] + s4;
#pragma unroll
        for (int nc = 0; nc < 4; ++nc) acc[nc][r] *= corr[r];
      }
      // O += P V  (P reloaded as A-fragments; V from transposed LDS tile)
      const v16h pf0 = load_afrag(Pw, LDK, 0, 0,  lane);
      const v16h pf1 = load_afrag(Pw, LDK, 0, 32, lane);
#pragma unroll
      for (int nc = 0; nc < 4; ++nc){
        acc[nc] = wmma16(pf0, load_bfrag(Vt, LDK, nc * 16, 0,  lane), acc[nc]);
        acc[nc] = wmma16(pf1, load_bfrag(Vt, LDK, nc * 16, 32, lane), acc[nc]);
      }
    }
  }
#pragma unroll
  for (int nc = 0; nc < 4; ++nc)
#pragma unroll
    for (int r = 0; r < 8; ++r)
      ctxh[(size_t)(rowQ + r + g * 8) * Hh + nc * 16 + n] = (_Float16)(acc[nc][r] / lrow[r]);
}

// ---------------- output projection: ctx[8192,64] @ Wsum[64,1024] + b ----------------

__global__ __launch_bounds__(256) void proj_kernel(const _Float16* __restrict__ ctxh,
                                                   const _Float16* __restrict__ wsum_t,
                                                   const float* __restrict__ bproj,
                                                   float* __restrict__ out){
  const int lane = threadIdx.x & 31;
  const int wave = threadIdx.x >> 5;
  const int gw = blockIdx.x * 8 + wave;   // 512 m-tiles * 64 n-tiles = 32768 waves
  const int mt = gw >> 6;
  const int nt = gw & 63;
  const int row = mt * 16, col = nt * 16;
  v8f c = {};
  c = wmma16(load_afrag(ctxh, Hh, row, 0,  lane), load_bfrag(wsum_t, Hh, col, 0,  lane), c);
  c = wmma16(load_afrag(ctxh, Hh, row, 32, lane), load_bfrag(wsum_t, Hh, col, 32, lane), c);
  const int n = lane & 15, g = lane >> 4;
  const float bias = bproj[col + n];
#pragma unroll
  for (int r = 0; r < 8; ++r)
    out[(size_t)(row + r + g * 8) * Dd + col + n] = c[r] + bias;
}

// ---------------- launch ----------------

extern "C" void kernel_launch(void* const* d_in, const int* in_sizes, int n_in,
                              void* d_out, int out_size, void* d_ws, size_t ws_size,
                              hipStream_t stream) {
  (void)in_sizes; (void)n_in; (void)out_size; (void)ws_size;
  const float* x  = (const float*)d_in[0];
  const float* Wq = (const float*)d_in[1];
  const float* Wk = (const float*)d_in[2];
  const float* Wv = (const float*)d_in[3];
  const float* Wp = (const float*)d_in[4];
  const float* bp = (const float*)d_in[5];
  float* out = (float*)d_out;

  _Float16* xh    = (_Float16*)d_ws;                       // 8192*1024
  _Float16* qh    = xh   + (size_t)NROWS * Dd;             // 8192*64
  _Float16* kh    = qh   + (size_t)NROWS * Hh;
  _Float16* vh    = kh   + (size_t)NROWS * Hh;
  _Float16* ctxh  = vh   + (size_t)NROWS * Hh;
  _Float16* wt3   = ctxh + (size_t)NROWS * Hh;             // 3*64*1024
  _Float16* wsum  = wt3  + (size_t)3 * Hh * Dd;            // 1024*64

  cvt_x_kernel<<<2048, 256, 0, stream>>>(x, xh);
  prep_w_kernel<<<256, 256, 0, stream>>>(Wq, Wk, Wv, Wp, wt3, wsum);
  qkv_kernel<<<768, 256, 0, stream>>>(xh, wt3, qh);
  attn_kernel<<<Bb * (Ss / 64), 128, 0, stream>>>(qh, kh, vh, ctxh);
  proj_kernel<<<4096, 256, 0, stream>>>(ctxh, wsum, bp, out);
}
